// GroupNet_18562848653351
// MI455X (gfx1250) — matrix operand order repaired and checked by reading, not compiled, not run on hardware
//
#include <hip/hip_runtime.h>
#include <hip/hip_bf16.h>

// ---- problem constants (match reference) ----
#define AG_N   2816                // agents
#define S_N    20                  // sample_num
#define B_N    (AG_N * S_N)        // 56320 rows
#define TP_N   10
#define TF_N   20
#define CH_N   32                  // conv channels
#define HG_N   96                  // GRU hidden
#define G3     (3 * HG_N)          // 288 gate outputs
#define DIN_N  1152                // 1024 + 32 + 96

typedef __attribute__((ext_vector_type(16))) _Float16 v16h;
typedef __attribute__((ext_vector_type(8)))  float    v8f;

__device__ __forceinline__ v8f wmma16(v16h a, v16h b, v8f c) {
  return __builtin_amdgcn_wmma_f32_16x16x32_f16(false, a, false, b, (short)0, c,
                                                false, false);
}

// A-matrix (16x32 f16) fragment from LDS f16 tile, row-major, leading dim ld.
// lanes 0-15: M=lane, K=0..7 & 16..23 ; lanes 16-31: K=8..15 & 24..31.
__device__ __forceinline__ v16h load_a_lds(const _Float16* tile, int ld, int k0,
                                           int lane) {
  const _Float16* p = tile + (lane & 15) * ld + k0 + ((lane >> 4) << 3);
  v16h a;
#pragma unroll
  for (int i = 0; i < 8; ++i) a[i] = p[i];
#pragma unroll
  for (int i = 0; i < 8; ++i) a[8 + i] = p[16 + i];
  return a;
}

// A-matrix fragment from LDS f32 tile (async-staged), converting to f16.
__device__ __forceinline__ v16h load_a_lds_f32(const float* tile, int ld, int k0,
                                               int lane) {
  const float* p = tile + (lane & 15) * ld + k0 + ((lane >> 4) << 3);
  v16h a;
#pragma unroll
  for (int i = 0; i < 8; ++i) a[i] = (_Float16)p[i];
#pragma unroll
  for (int i = 0; i < 8; ++i) a[8 + i] = (_Float16)p[16 + i];
  return a;
}

// B-matrix (32x16 f16) fragment from LDS; weights row-major (N rows x K cols).
__device__ __forceinline__ v16h load_b_lds(const _Float16* w, int ldk, int n0,
                                           int k0, int lane) {
  const _Float16* p = w + (n0 + (lane & 15)) * ldk + k0 + ((lane >> 4) << 3);
  v16h b;
#pragma unroll
  for (int i = 0; i < 8; ++i) b[i] = p[i];
#pragma unroll
  for (int i = 0; i < 8; ++i) b[8 + i] = p[16 + i];
  return b;
}

// B-matrix fragment straight from global f32 weights (row-major N x K), cvt f16.
__device__ __forceinline__ v16h load_b_glb(const float* __restrict__ w, int ldk,
                                           int n0, int k0, int lane, int nmax) {
  int n = n0 + (lane & 15);
  if (n >= nmax) n = nmax - 1;  // clamp to avoid OOB on padded col blocks
  const float* p = w + (size_t)n * ldk + k0 + ((lane >> 4) << 3);
  v16h b;
#pragma unroll
  for (int i = 0; i < 8; ++i) b[i] = (_Float16)p[i];
#pragma unroll
  for (int i = 0; i < 8; ++i) b[8 + i] = (_Float16)p[16 + i];
  return b;
}

// fast transcendentals: v_rcp_f32 + v_exp_f32 (avoid div_scale/div_fixup chains)
__device__ __forceinline__ float sigmf(float x) {
  return __builtin_amdgcn_rcpf(1.f + __expf(-x));
}
__device__ __forceinline__ float tanh_fast(float x) {
  return 2.f * __builtin_amdgcn_rcpf(1.f + __expf(-2.f * x)) - 1.f;
}

// gfx1250 async global->LDS copy (16B per lane), ASYNCcnt-tracked.
__device__ __forceinline__ void async_copy_b128(uint32_t lds_off, uint32_t g_off,
                                                const void* base) {
  asm volatile("global_load_async_to_lds_b128 %0, %1, %2"
               :: "v"(lds_off), "v"(g_off), "s"(base)
               : "memory");
}
__device__ __forceinline__ void wait_async0() {
  asm volatile("s_wait_asynccnt 0x0" ::: "memory");
}

// --------------------------------------------------------------------------
__global__ void zero_kernel(float* __restrict__ p, int n) {
  int i = blockIdx.x * blockDim.x + threadIdx.x;
  if (i < n) p[i] = 0.f;
}

// --------------------------------------------------------------------------
// conv1d(2->32,k=3,pad=1) + relu + 10-step GRU(32->96) via WMMA.
// 128 threads = 4 waves; each wave owns a 16-row tile. grid = B_N/64.
__global__ __launch_bounds__(128, 1) void gru_embed_kernel(
    const float* __restrict__ past_traj, const float* __restrict__ xhat,
    const float* __restrict__ conv_w, const float* __restrict__ conv_b,
    const float* __restrict__ wih, const float* __restrict__ whh,
    const float* __restrict__ bih, const float* __restrict__ bhh,
    float* __restrict__ state) {
  extern __shared__ char smem[];
  _Float16* wih_s = (_Float16*)smem;            // 288*32 f16
  _Float16* whh_s = wih_s + G3 * CH_N;          // 288*96 f16
  float* bih_s = (float*)(whh_s + G3 * HG_N);   // 288 f32
  float* bhh_s = bih_s + G3;                    // 288 f32
  char* pwave = (char*)(bhh_s + G3);
  const int tid = threadIdx.x, lane = tid & 31, wave = tid >> 5;

  for (int i = tid; i < G3 * CH_N; i += 128) wih_s[i] = (_Float16)wih[i];
  for (int i = tid; i < G3 * HG_N; i += 128) whh_s[i] = (_Float16)whh[i];
  for (int i = tid; i < G3; i += 128) { bih_s[i] = bih[i]; bhh_s[i] = bhh[i]; }

  const int WB = (TP_N * 16 * CH_N + 16 * HG_N) * 2 + 16 * TP_N * 2 * 4; // 14592B
  char* base = pwave + wave * WB;
  _Float16* emb  = (_Float16*)base;              // [10][16][32] f16
  _Float16* hbuf = emb + TP_N * 16 * CH_N;       // [16][96]  f16
  float*    dloc = (float*)(hbuf + 16 * HG_N);   // [16][20]  f32

  const int row0 = blockIdx.x * 64 + wave * 16;

  for (int e = lane; e < 16 * 20; e += 32) {
    int r = e / 20, c = e % 20;
    int rg = row0 + r;
    int ag = rg / S_N;
    dloc[e] = past_traj[ag * 20 + c] - xhat[(size_t)rg * 20 + c];
  }
  for (int e = lane; e < 16 * HG_N; e += 32) hbuf[e] = (_Float16)0.f;
  __syncthreads();

  // conv: lane == channel
  float cw[6];
  const float cb0 = conv_b[lane];
#pragma unroll
  for (int q = 0; q < 6; ++q) cw[q] = conv_w[lane * 6 + q];
  for (int r = 0; r < 16; ++r) {
    const float* dr = dloc + r * 20;
#pragma unroll
    for (int t = 0; t < TP_N; ++t) {
      float acc = cb0;
#pragma unroll
      for (int kt = 0; kt < 3; ++kt) {
        int tt = t + kt - 1;
        if (tt >= 0 && tt < TP_N)
          acc += cw[kt] * dr[tt * 2] + cw[3 + kt] * dr[tt * 2 + 1];
      }
      emb[(t * 16 + r) * CH_N + lane] = (_Float16)fmaxf(acc, 0.f);
    }
  }
  __syncthreads();

  // GRU: gates r,z,n ; 72 WMMAs per step per wave
  for (int t = 0; t < TP_N; ++t) {
    v16h Ax  = load_a_lds(emb + t * 16 * CH_N, CH_N, 0, lane);
    v16h Ah0 = load_a_lds(hbuf, HG_N, 0, lane);
    v16h Ah1 = load_a_lds(hbuf, HG_N, 32, lane);
    v16h Ah2 = load_a_lds(hbuf, HG_N, 64, lane);
#pragma unroll
    for (int j = 0; j < 6; ++j) {
      const int nr = j * 16, nzg = 96 + j * 16, nng = 192 + j * 16;
      v8f Cr = {}, Cz = {}, Cin = {}, Chn = {};
      Cr  = wmma16(Ax,  load_b_lds(wih_s, CH_N, nr, 0, lane), Cr);
      Cr  = wmma16(Ah0, load_b_lds(whh_s, HG_N, nr, 0, lane), Cr);
      Cr  = wmma16(Ah1, load_b_lds(whh_s, HG_N, nr, 32, lane), Cr);
      Cr  = wmma16(Ah2, load_b_lds(whh_s, HG_N, nr, 64, lane), Cr);
      Cz  = wmma16(Ax,  load_b_lds(wih_s, CH_N, nzg, 0, lane), Cz);
      Cz  = wmma16(Ah0, load_b_lds(whh_s, HG_N, nzg, 0, lane), Cz);
      Cz  = wmma16(Ah1, load_b_lds(whh_s, HG_N, nzg, 32, lane), Cz);
      Cz  = wmma16(Ah2, load_b_lds(whh_s, HG_N, nzg, 64, lane), Cz);
      Cin = wmma16(Ax,  load_b_lds(wih_s, CH_N, nng, 0, lane), Cin);
      Chn = wmma16(Ah0, load_b_lds(whh_s, HG_N, nng, 0, lane), Chn);
      Chn = wmma16(Ah1, load_b_lds(whh_s, HG_N, nng, 32, lane), Chn);
      Chn = wmma16(Ah2, load_b_lds(whh_s, HG_N, nng, 64, lane), Chn);
      const int ncol = j * 16 + (lane & 15);
      const float br   = bih_s[ncol] + bhh_s[ncol];
      const float bz   = bih_s[96 + ncol] + bhh_s[96 + ncol];
      const float bin_ = bih_s[192 + ncol];
      const float bhn  = bhh_s[192 + ncol];
#pragma unroll
      for (int jj = 0; jj < 8; ++jj) {
        int m = jj + ((lane >> 4) << 3);  // C-layout row within 16x16 tile
        float rv = sigmf(Cr[jj] + br);
        float zv = sigmf(Cz[jj] + bz);
        float nv = tanh_fast(Cin[jj] + bin_ + rv * (Chn[jj] + bhn));
        float hold = (float)hbuf[m * HG_N + ncol];
        hbuf[m * HG_N + ncol] = (_Float16)((1.f - zv) * nv + zv * hold);
      }
    }
    __syncthreads();
  }
  for (int e = lane; e < 16 * HG_N; e += 32) {
    int r = e / HG_N, c = e % HG_N;
    state[(size_t)(row0 + r) * HG_N + c] = (float)hbuf[e];
  }
}

// --------------------------------------------------------------------------
// Issue async copies for one 64x32 f32 feat chunk into an LDS buffer.
// The concat [pf | z | state] is column ranges; each chunk maps to one source.
__device__ __forceinline__ void issue_feat_async(
    const float* __restrict__ pf, const float* __restrict__ zin,
    const float* __restrict__ state, int row0, int kc, float* dstbuf, int tid) {
  const float* base;
  uint32_t rstride, coff;
  const int k0 = kc * 32;
  if (k0 < 1024)      { base = pf;    rstride = 4096; coff = (uint32_t)k0 * 4; }
  else if (k0 < 1056) { base = zin;   rstride = 128;  coff = 0; }
  else                { base = state; rstride = 384;  coff = (uint32_t)(k0 - 1056) * 4; }
  const uint32_t dstbase = (uint32_t)(uintptr_t)dstbuf;
  // 64 rows * 128B = 512 segments of 16B; 256 threads -> 2 async ops per thread
#pragma unroll
  for (int e = tid; e < 512; e += 256) {
    uint32_t r = (uint32_t)e >> 3, seg = (uint32_t)e & 7;
    uint32_t ldsoff = dstbase + r * 128u + seg * 16u;
    uint32_t goff = (uint32_t)(row0 + (int)r) * rstride + coff + seg * 16u;
    async_copy_b128(ldsoff, goff, base);
  }
}

// --------------------------------------------------------------------------
// Fused MLP 1152->512->256->nout on a 64-row tile. 256 threads = 8 waves.
// mode 0: write xhat, accumulate recon (stride 20). mode 1: accumulate pred (40).
__global__ __launch_bounds__(256, 1) void mlp_kernel(
    const float* __restrict__ pf, const float* __restrict__ zin,
    const float* __restrict__ state,
    const float* __restrict__ w1, const float* __restrict__ b1,
    const float* __restrict__ w2, const float* __restrict__ b2,
    const float* __restrict__ w3, const float* __restrict__ b3,
    int nout, float* __restrict__ xhat, float* __restrict__ accum, int mode) {
  extern __shared__ char smem[];
  float* fb0 = (float*)smem;             // 64*32 f32 (async stage, buf 0)
  float* fb1 = fb0 + 64 * 32;            // 64*32 f32 (buf 1)
  _Float16* h1b = (_Float16*)(fb1 + 64 * 32);  // 64*512 f16
  _Float16* h2b = h1b + 64 * 512;              // 64*256 f16
  const int tid = threadIdx.x, lane = tid & 31, wave = tid >> 5;
  const int row0 = blockIdx.x * 64;
  float* fb[2] = {fb0, fb1};

  // ---- layer 1: N=512, K=1152, async double-buffered A staging ----
  v8f zc = {};
  v8f acc1[4][4];
#pragma unroll
  for (int rb = 0; rb < 4; ++rb)
#pragma unroll
    for (int cb = 0; cb < 4; ++cb) acc1[rb][cb] = zc;
  const int n0w = wave * 64;

  issue_feat_async(pf, zin, state, row0, 0, fb[0], tid);
  for (int kc = 0; kc < DIN_N / 32; ++kc) {
    const int k0 = kc * 32;
    wait_async0();
    __syncthreads();  // all waves' async copies landed in LDS
    const float* cur = fb[kc & 1];
    if (kc + 1 < DIN_N / 32)
      issue_feat_async(pf, zin, state, row0, kc + 1, fb[(kc + 1) & 1], tid);
    // prefetch next weight chunk into caches (global_prefetch_b8)
    if (kc + 1 < DIN_N / 32)
      __builtin_prefetch(w1 + (size_t)(n0w + (lane & 15)) * DIN_N + k0 + 32, 0, 1);
    v16h bf[4];
#pragma unroll
    for (int cb = 0; cb < 4; ++cb)
      bf[cb] = load_b_glb(w1, DIN_N, n0w + cb * 16, k0, lane, 512);
    v16h af[4];
#pragma unroll
    for (int rb = 0; rb < 4; ++rb)
      af[rb] = load_a_lds_f32(cur + rb * 16 * 32, 32, 0, lane);
#pragma unroll
    for (int rb = 0; rb < 4; ++rb)
#pragma unroll
      for (int cb = 0; cb < 4; ++cb)
        acc1[rb][cb] = wmma16(af[rb], bf[cb], acc1[rb][cb]);
  }
  __syncthreads();
#pragma unroll
  for (int rb = 0; rb < 4; ++rb)
#pragma unroll
    for (int cb = 0; cb < 4; ++cb) {
      int n = n0w + cb * 16 + (lane & 15);
      float bias = b1[n];
#pragma unroll
      for (int jj = 0; jj < 8; ++jj) {
        int m = rb * 16 + jj + ((lane >> 4) << 3);
        h1b[m * 512 + n] = (_Float16)fmaxf(acc1[rb][cb][jj] + bias, 0.f);
      }
    }
  __syncthreads();

  // ---- layer 2: N=256, K=512 ----
  v8f acc2[4][2];
#pragma unroll
  for (int rb = 0; rb < 4; ++rb)
#pragma unroll
    for (int cb = 0; cb < 2; ++cb) acc2[rb][cb] = zc;
  const int n0w2 = wave * 32;
  for (int kc = 0; kc < 16; ++kc) {
    const int k0 = kc * 32;
    v16h bf2[2];
#pragma unroll
    for (int cb = 0; cb < 2; ++cb)
      bf2[cb] = load_b_glb(w2, 512, n0w2 + cb * 16, k0, lane, 256);
    v16h af2[4];
#pragma unroll
    for (int rb = 0; rb < 4; ++rb)
      af2[rb] = load_a_lds(h1b + rb * 16 * 512, 512, k0, lane);
#pragma unroll
    for (int rb = 0; rb < 4; ++rb)
#pragma unroll
      for (int cb = 0; cb < 2; ++cb)
        acc2[rb][cb] = wmma16(af2[rb], bf2[cb], acc2[rb][cb]);
  }
#pragma unroll
  for (int rb = 0; rb < 4; ++rb)
#pragma unroll
    for (int cb = 0; cb < 2; ++cb) {
      int n = n0w2 + cb * 16 + (lane & 15);
      float bias = b2[n];
#pragma unroll
      for (int jj = 0; jj < 8; ++jj) {
        int m = rb * 16 + jj + ((lane >> 4) << 3);
        h2b[m * 256 + n] = (_Float16)fmaxf(acc2[rb][cb][jj] + bias, 0.f);
      }
    }
  __syncthreads();

  // ---- layer 3: N=nout (20 or 40, padded to 16-blocks), K=256 ----
  if (wave < 4) {
    const int rb = wave;
    const int nblk = (nout + 15) >> 4;
    for (int cb = 0; cb < nblk; ++cb) {
      v8f c3 = zc;
      for (int kc = 0; kc < 8; ++kc) {
        v16h a3 = load_a_lds(h2b + rb * 16 * 256, 256, kc * 32, lane);
        v16h b3f = load_b_glb(w3, 256, cb * 16, kc * 32, lane, nout);
        c3 = wmma16(a3, b3f, c3);
      }
      int n = cb * 16 + (lane & 15);
      if (n < nout) {
        float bias = b3[n];
#pragma unroll
        for (int jj = 0; jj < 8; ++jj) {
          size_t rg = (size_t)(row0 + rb * 16 + jj + ((lane >> 4) << 3));
          float val = c3[jj] + bias;
          if (mode == 0) {
            xhat[rg * 20 + n] = val;     // feeds next decompose iteration
            accum[rg * 20 + n] += val;   // reconstruction accumulator
          } else {
            accum[rg * 40 + n] += val;   // prediction accumulator
          }
        }
      }
    }
  }
}

// --------------------------------------------------------------------------
__global__ void finalize_kernel(const float* __restrict__ pred,
                                const float* __restrict__ recon,
                                const float* __restrict__ cur,
                                float* __restrict__ out) {
  int idx = blockIdx.x * blockDim.x + threadIdx.x;
  const int nA = B_N * TF_N * 2;  // out_seq elements
  if (idx < nA) {
    int row = idx / 40;
    int ax = idx & 1;
    int ag = row / S_N;
    out[idx] = pred[idx] + cur[ag * 2 + ax];
  } else if (idx < nA + B_N * TP_N * 2) {
    out[idx] = recon[idx - nA];
  }
}

// --------------------------------------------------------------------------
extern "C" void kernel_launch(void* const* d_in, const int* in_sizes, int n_in,
                              void* d_out, int out_size, void* d_ws,
                              size_t ws_size, hipStream_t stream) {
  (void)in_sizes; (void)n_in; (void)out_size; (void)ws_size;
  const float* pf        = (const float*)d_in[0];
  const float* zin       = (const float*)d_in[1];
  const float* past_traj = (const float*)d_in[2];
  const float* cur       = (const float*)d_in[3];
  const float* conv_w    = (const float*)d_in[4];
  const float* conv_b    = (const float*)d_in[5];
  const float* gwih      = (const float*)d_in[6];
  const float* gwhh      = (const float*)d_in[7];
  const float* gbih      = (const float*)d_in[8];
  const float* gbhh      = (const float*)d_in[9];
  const float* dxw1 = (const float*)d_in[10];  const float* dxb1 = (const float*)d_in[11];
  const float* dxw2 = (const float*)d_in[12];  const float* dxb2 = (const float*)d_in[13];
  const float* dxw3 = (const float*)d_in[14];  const float* dxb3 = (const float*)d_in[15];
  const float* dyw1 = (const float*)d_in[16];  const float* dyb1 = (const float*)d_in[17];
  const float* dyw2 = (const float*)d_in[18];  const float* dyb2 = (const float*)d_in[19];
  const float* dyw3 = (const float*)d_in[20];  const float* dyb3 = (const float*)d_in[21];

  float* ws    = (float*)d_ws;
  float* xhat  = ws;                       // B*20
  float* recon = xhat + (size_t)B_N * 20;  // B*20
  float* pred  = recon + (size_t)B_N * 20; // B*40
  float* state = pred + (size_t)B_N * 40;  // B*96

  const int nz = B_N * 80;  // xhat + recon + pred contiguous
  zero_kernel<<<(nz + 255) / 256, 256, 0, stream>>>(ws, nz);

  const size_t gru_sh = (size_t)(G3 * CH_N + G3 * HG_N) * 2 + (size_t)G3 * 2 * 4 +
                        4 * ((size_t)(TP_N * 16 * CH_N + 16 * HG_N) * 2 +
                             (size_t)16 * TP_N * 2 * 4);          // 134400 B
  const size_t mlp_sh = (size_t)(2 * 64 * 32) * 4 +
                        (size_t)(64 * 512 + 64 * 256) * 2;        // 114688 B
  const int grid = B_N / 64;  // 880

  for (int i = 0; i < 2; ++i) {
    gru_embed_kernel<<<grid, 128, gru_sh, stream>>>(
        past_traj, xhat, conv_w + i * CH_N * 6, conv_b + i * CH_N,
        gwih + i * G3 * CH_N, gwhh + i * G3 * HG_N, gbih + i * G3,
        gbhh + i * G3, state);
    mlp_kernel<<<grid, 256, mlp_sh, stream>>>(
        pf, zin, state, dxw1 + (size_t)i * 512 * DIN_N, dxb1 + i * 512,
        dxw2 + (size_t)i * 256 * 512, dxb2 + i * 256,
        dxw3 + (size_t)i * 20 * 256, dxb3 + i * 20, 20, xhat, recon, 0);
    mlp_kernel<<<grid, 256, mlp_sh, stream>>>(
        pf, zin, state, dyw1 + (size_t)i * 512 * DIN_N, dyb1 + i * 512,
        dyw2 + (size_t)i * 256 * 512, dyb2 + i * 256,
        dyw3 + (size_t)i * 40 * 256, dyb3 + i * 40, 40, nullptr, pred, 1);
  }
  const int nf = B_N * 60;
  finalize_kernel<<<(nf + 255) / 256, 256, 0, stream>>>(pred, recon, cur,
                                                        (float*)d_out);
}
